// ModelPositional_75136157876562
// MI455X (gfx1250) — compile-verified, matrix operand order checked by reading, not run on hardware
//
#include <hip/hip_runtime.h>
#include <hip/hip_bf16.h>

// ---------------------------------------------------------------------------
// Shapes
// ---------------------------------------------------------------------------
#define BB      4
#define SS      512
#define KPE     16
#define VOCAB   50265
#define DEMB    768
#define DCAT    784          // 768 + 16
#define KPAD    800          // DCAT padded to multiple of 32
#define ROWS    (BB * SS)    // 2048
#define LSTR    40           // bf16 elems per LDS row; 80B pitch (16B aligned, padded)

typedef __attribute__((ext_vector_type(16))) __bf16 v16bf;
typedef __attribute__((ext_vector_type(8)))  __bf16 v8bf;
typedef __attribute__((ext_vector_type(8)))  float  v8f;
typedef __attribute__((ext_vector_type(4)))  unsigned int v4u;
typedef __attribute__((ext_vector_type(8)))  int    v8i;
typedef __attribute__((ext_vector_type(4)))  int    v4i;

#ifndef __has_builtin
#define __has_builtin(x) 0
#endif
#if __has_builtin(__builtin_amdgcn_tensor_load_to_lds) && \
    __has_builtin(__builtin_amdgcn_s_wait_tensorcnt)
#define HAVE_TDM 1
#else
#define HAVE_TDM 0
#endif

#if HAVE_TDM
// ---------------------------------------------------------------------------
// TDM: async DMA of a 128-row x 32-elem bf16 tile (row stride = stride_elems)
// from global into LDS at byte offset lds_off, padding each 64B row to the
// 80B LDS pitch (LSTR = 40 bf16). One instruction per wave; TENSORcnt-tracked.
// D# layout per CDNA5 ISA ch.8 (group0: count/lds/global/type; group1: dims).
// ---------------------------------------------------------------------------
__device__ __forceinline__ void tdm_load_tile_bf16(unsigned lds_off,
                                                   const void* gptr,
                                                   long stride_elems)
{
    const unsigned long long ga = (unsigned long long)gptr;
    const v4u g0 = {
        1u,                                              // count=1, user mode
        lds_off,                                         // LDS byte address
        (unsigned)ga,                                    // global_addr[31:0]
        (unsigned)((ga >> 32) & 0x1FFFFFFull) | (2u << 30) // [56:32] | type=2
    };
    const v8i g1 = {
        (int)((1u << 16)      // data_size = 1 -> 2 bytes
            | (1u << 20)      // pad_enable
            | (3u << 22)      // pad_interval = 3 -> after 16 dwords (64B)
            | (3u << 25)),    // pad_amount   = 3 -> 4 dwords (16B) => 80B pitch
        (int)(32u << 16),     // atomic_barrier_addr=0 | tensor_dim0 = 32
        (int)(128u << 16),    // tensor_dim0 hi = 0    | tensor_dim1 = 128
        (int)(32u << 16),     // tensor_dim1 hi = 0    | tile_dim0 = 32
        (int)128,             // tile_dim1 = 128       | tile_dim2 = 0
        (int)(unsigned)stride_elems,   // tensor_dim0_stride[31:0]
        0,                    // stride hi = 0 | tensor_dim1_stride lo = 0
        0
    };
    const v4i gz = {0, 0, 0, 0};
#if __clang_major__ >= 23
    const v8i gz8 = {0, 0, 0, 0, 0, 0, 0, 0};
    __builtin_amdgcn_tensor_load_to_lds(g0, g1, gz, gz, gz8, 0);
#else
    __builtin_amdgcn_tensor_load_to_lds(g0, g1, gz, gz, 0);
#endif
}
#endif

// ---------------------------------------------------------------------------
// Kernel 1: M = D^-1 A (bf16, row-major) and Mt = M^T (bf16, for the WMMA
// B-operand: pure-copy staging). Also pe col 0 = diag(M) = 1/deg into X (bf16).
// ---------------------------------------------------------------------------
__global__ __launch_bounds__(256)
void build_m_kernel(const unsigned char* __restrict__ attn,
                    __bf16* __restrict__ Mbf,
                    __bf16* __restrict__ Mtbf,
                    __bf16* __restrict__ diag0 /* = X + 768, row stride KPAD */)
{
    const int bi = blockIdx.x;           // b*512 + i
    const int b  = bi >> 9;
    const int i  = bi & (SS - 1);
    const int t  = threadIdx.x;
    const unsigned char* arow = attn + (long)bi * SS;

    const float a0 = (t == i)       ? 1.f : (arow[t]       ? 1.f : 0.f);
    const float a1 = (t + 256 == i) ? 1.f : (arow[t + 256] ? 1.f : 0.f);

    __shared__ float red[256];
    red[t] = a0 + a1;
    __syncthreads();
    #pragma unroll
    for (int off = 128; off >= 1; off >>= 1) {
        if (t < off) red[t] += red[t + off];
        __syncthreads();
    }
    const float deg = red[0];
    const float inv = deg > 0.f ? 1.f / deg : 0.f;

    const __bf16 m0v = (__bf16)(a0 * inv);
    const __bf16 m1v = (__bf16)(a1 * inv);
    __bf16* mrow = Mbf + (long)bi * SS;
    mrow[t]       = m0v;
    mrow[t + 256] = m1v;
    __bf16* mt = Mtbf + (long)b * SS * SS;          // Mt[j][i] = M[i][j]
    mt[(long)t * SS + i]         = m0v;
    mt[(long)(t + 256) * SS + i] = m1v;
    if (t == 0) diag0[(long)bi * KPAD] = (__bf16)inv;
}

// ---------------------------------------------------------------------------
// Kernel 2: embedding gather + masked node averaging -> X (bf16) cols [0,768),
// zero-fill K padding cols [784,800).
// ---------------------------------------------------------------------------
__global__ __launch_bounds__(256)
void emb_avg_kernel(const int* __restrict__ code,
                    const int* __restrict__ pos,
                    const unsigned char* __restrict__ attn,
                    const float* __restrict__ emb_table,
                    __bf16* __restrict__ X)
{
    const int bi = blockIdx.x;           // b*512 + i
    const int b  = bi >> 9;
    const int t  = threadIdx.x;
    __bf16* xrow = X + (long)bi * KPAD;

    __shared__ int scode[SS];

    const int my_pos = pos[bi];          // uniform across block
    if (my_pos != 0) {
        const float* e = emb_table + (long)code[bi] * DEMB;
        xrow[t]       = (__bf16)e[t];
        xrow[t + 256] = (__bf16)e[t + 256];
        xrow[t + 512] = (__bf16)e[t + 512];
    } else {
        const unsigned char* arow = attn + (long)bi * SS;
        const int base = b << 9;
        #pragma unroll
        for (int r = 0; r < 2; ++r) {
            const int j = t + (r << 8);
            scode[j] = (pos[base + j] >= 2 && arow[j]) ? code[base + j] : -1;
        }
        __syncthreads();
        float a0 = 0.f, a1 = 0.f, a2 = 0.f, w = 0.f;
        for (int j = 0; j < SS; ++j) {
            const int c = scode[j];
            if (c >= 0) {
                const float* e = emb_table + (long)c * DEMB + t;
                w += 1.f;
                a0 += e[0]; a1 += e[256]; a2 += e[512];
            }
        }
        const float inv = 1.f / (w + 1e-10f);
        xrow[t]       = (__bf16)(a0 * inv);
        xrow[t + 256] = (__bf16)(a1 * inv);
        xrow[t + 512] = (__bf16)(a2 * inv);
    }
    if (t < KPAD - DCAT) xrow[DCAT + t] = (__bf16)0.f;
}

// ---------------------------------------------------------------------------
// B-operand register staging (overloaded on element type).
//   bf16: B is PRE-TRANSPOSED (rows = n, contiguous k) -> pure 32B copy.
//   f32 : B is k-major (W) -> strided gather + cvt; block-uniform fast path.
// ---------------------------------------------------------------------------
__device__ __forceinline__ void loadB_regs(const __bf16* __restrict__ Bm, long ldb,
        int n0, int idx, int kH, int kb, int N, int Kvalid, v8bf& lo, v8bf& hi)
{
    const __bf16* s = Bm + (long)(n0 + idx) * ldb + kb + kH;
    lo = *(const v8bf*)s;
    hi = *(const v8bf*)(s + 8);
}

__device__ __forceinline__ void loadB_regs(const float* __restrict__ Bm, long ldb,
        int n0, int idx, int kH, int kb, int N, int Kvalid, v8bf& lo, v8bf& hi)
{
    float f[16];
    const float* s = Bm + (long)(kb + kH) * ldb + (n0 + idx);
    if ((n0 + 128 <= N) && (kb + 32 <= Kvalid)) {       // block-uniform fast path
        #pragma unroll
        for (int kk = 0; kk < 16; ++kk) { f[kk] = *s; s += ldb; }
    } else {
        const bool nv = (n0 + idx) < N;
        #pragma unroll
        for (int kk = 0; kk < 16; ++kk) {
            f[kk] = (nv && (kb + kH + kk) < Kvalid) ? *s : 0.f;
            s += ldb;
        }
    }
    #pragma unroll
    for (int kk = 0; kk < 8; ++kk) {
        lo[kk] = (__bf16)f[kk];
        hi[kk] = (__bf16)f[kk + 8];
    }
}

__device__ __forceinline__ void prefetchB(const float* __restrict__ Bm, long ldb,
        int n0, int idx, int kH, int kb)
{
    __builtin_prefetch(Bm + (long)(kb + kH) * ldb + (n0 + idx), 0, 0);
}
__device__ __forceinline__ void prefetchB(const __bf16*, long, int, int, int, int) {}

// ---------------------------------------------------------------------------
// Kernel 3: bf16 WMMA GEMM  C = A(bf16) * B [+ bias]
//   block tile 128x128, BLK_K=32, double-buffered LDS, one barrier per K-step.
//   8 waves = 4(M)x2(N); each wave 32x64 = 2x4 WMMA tiles.
//   bf16 tile staging via TDM (tensor_load_to_lds, TENSORcnt) when available;
//   f32 W staging via registers + cvt (+ global_prefetch).
//   Optional fused diagonal scatter (random-walk PE chain).
// ---------------------------------------------------------------------------
template <typename BT, typename CT>
__global__ __launch_bounds__(256)
void wmma_gemm_kernel(const __bf16* __restrict__ A, long lda, long bsA,
                      const BT* __restrict__ Bm, long ldb, long bsB,
                      CT* __restrict__ C, long ldc, long bsC,
                      int N, int Kdim, int Kvalid,
                      const float* __restrict__ bias,
                      __bf16* __restrict__ diag_out, long diag_ld, long diag_bs)
{
    __shared__ __attribute__((aligned(16))) __bf16 ldsA[2][128 * LSTR];
    __shared__ __attribute__((aligned(16))) __bf16 ldsB[2][128 * LSTR];

    constexpr bool B_IS_BF16 = (sizeof(BT) == 2);

    const int t    = threadIdx.x;
    const int lane = t & 31;
    const int wave = t >> 5;
    const int wm   = wave & 3;       // 4 waves along M (32 rows each)
    const int wn   = wave >> 2;      // 2 waves along N (64 cols each)
    const int zb   = blockIdx.z;
    const int m0   = blockIdx.x * 128;
    const int n0   = blockIdx.y * 128;

    A  += (long)zb * bsA;
    Bm += (long)zb * bsB;
    C  += (long)zb * bsC;

    const int idx = t & 127;             // staging row index (m or n)
    const int kH  = (t >> 7) << 4;       // staging k sub-block: 0 or 16
    const int rr  = lane & 15;           // row/col within 16x16 tile
    const int hh  = lane >> 4;           // K-half selector (WMMA 16-bit layout)
    const unsigned lofs = idx * LSTR + kH;

    v8f acc[2][4] = {};
    v8bf alo, ahi, blo, bhi;

    // ---- prologue: stage k-block 0 into buffer 0 ----
#if HAVE_TDM
    if (wave == 0)
        tdm_load_tile_bf16((unsigned)(unsigned long long)&ldsA[0][0],
                           A + (long)m0 * lda, lda);
    if constexpr (B_IS_BF16) {
        if (wave == 1)
            tdm_load_tile_bf16((unsigned)(unsigned long long)&ldsB[0][0],
                               (const __bf16*)Bm + (long)n0 * ldb, ldb);
    } else {
        loadB_regs(Bm, ldb, n0, idx, kH, 0, N, Kvalid, blo, bhi);
        *(v8bf*)(&ldsB[0][lofs])     = blo;
        *(v8bf*)(&ldsB[0][lofs + 8]) = bhi;
    }
    __builtin_amdgcn_s_wait_tensorcnt(0);
#else
    {
        const __bf16* s = A + (long)(m0 + idx) * lda + kH;
        alo = *(const v8bf*)s;
        ahi = *(const v8bf*)(s + 8);
        loadB_regs(Bm, ldb, n0, idx, kH, 0, N, Kvalid, blo, bhi);
        *(v8bf*)(&ldsA[0][lofs])     = alo;
        *(v8bf*)(&ldsA[0][lofs + 8]) = ahi;
        *(v8bf*)(&ldsB[0][lofs])     = blo;
        *(v8bf*)(&ldsB[0][lofs + 8]) = bhi;
    }
#endif
    __syncthreads();

    int cur = 0;
    for (int kb = 0; kb < Kdim; kb += 32) {
        const bool hasNext = (kb + 32) < Kdim;
        const int nxt = cur ^ 1;
        if (hasNext) {
#if HAVE_TDM
            if (wave == 0)
                tdm_load_tile_bf16((unsigned)(unsigned long long)&ldsA[nxt][0],
                                   A + (long)m0 * lda + (kb + 32), lda);
            if constexpr (B_IS_BF16) {
                if (wave == 1)
                    tdm_load_tile_bf16((unsigned)(unsigned long long)&ldsB[nxt][0],
                                       (const __bf16*)Bm + (long)n0 * ldb + (kb + 32), ldb);
            } else {
                loadB_regs(Bm, ldb, n0, idx, kH, kb + 32, N, Kvalid, blo, bhi);
                if ((kb + 96 <= Kvalid) && (n0 + 128 <= N))
                    prefetchB(Bm, ldb, n0, idx, kH, kb + 64);   // L2 prefetch of W
            }
#else
            {
                const __bf16* s = A + (long)(m0 + idx) * lda + (kb + 32) + kH;
                alo = *(const v8bf*)s;
                ahi = *(const v8bf*)(s + 8);
            }
            loadB_regs(Bm, ldb, n0, idx, kH, kb + 32, N, Kvalid, blo, bhi);
            if ((kb + 96 <= Kvalid) && (n0 + 128 <= N))
                prefetchB(Bm, ldb, n0, idx, kH, kb + 64);
#endif
        }

        // Fragments from LDS. ISA 16-bit A/B layout: lane%16 = row/col,
        // lane/16 = K half; elems 0..7 -> K=8h+e, elems 8..15 -> K=16+8h+(e-8).
        v16bf af[2], bfr[4];
        #pragma unroll
        for (int mi = 0; mi < 2; ++mi) {
            const __bf16* p = &ldsA[cur][(wm * 32 + mi * 16 + rr) * LSTR + hh * 8];
            const v8bf x0 = *(const v8bf*)p;
            const v8bf x1 = *(const v8bf*)(p + 16);
            af[mi] = __builtin_shufflevector(x0, x1,
                         0,1,2,3,4,5,6,7,8,9,10,11,12,13,14,15);
        }
        #pragma unroll
        for (int ni = 0; ni < 4; ++ni) {
            const __bf16* p = &ldsB[cur][(wn * 64 + ni * 16 + rr) * LSTR + hh * 8];
            const v8bf x0 = *(const v8bf*)p;
            const v8bf x1 = *(const v8bf*)(p + 16);
            bfr[ni] = __builtin_shufflevector(x0, x1,
                         0,1,2,3,4,5,6,7,8,9,10,11,12,13,14,15);
        }
        #pragma unroll
        for (int mi = 0; mi < 2; ++mi)
            #pragma unroll
            for (int ni = 0; ni < 4; ++ni)
                acc[mi][ni] = __builtin_amdgcn_wmma_f32_16x16x32_bf16(
                    false, af[mi], false, bfr[ni],
                    (short)0, acc[mi][ni], false, false);

        if (hasNext) {
#if HAVE_TDM
            if constexpr (!B_IS_BF16) {
                *(v8bf*)(&ldsB[nxt][lofs])     = blo;
                *(v8bf*)(&ldsB[nxt][lofs + 8]) = bhi;
            }
            __builtin_amdgcn_s_wait_tensorcnt(0);
#else
            *(v8bf*)(&ldsA[nxt][lofs])     = alo;
            *(v8bf*)(&ldsA[nxt][lofs + 8]) = ahi;
            *(v8bf*)(&ldsB[nxt][lofs])     = blo;
            *(v8bf*)(&ldsB[nxt][lofs + 8]) = bhi;
#endif
            __syncthreads();
            cur = nxt;
        }
    }

    // Epilogue: C/D layout = VGPR r -> row r+8h, lane%16 -> col.
    #pragma unroll
    for (int mi = 0; mi < 2; ++mi) {
        #pragma unroll
        for (int ni = 0; ni < 4; ++ni) {
            const int col = n0 + wn * 64 + ni * 16 + rr;
            if (col < N) {
                const float bv = bias ? bias[col] : 0.f;
                #pragma unroll
                for (int r = 0; r < 8; ++r) {
                    const int row = m0 + wm * 32 + mi * 16 + r + hh * 8;
                    C[(long)row * ldc + col] = (CT)(acc[mi][ni][r] + bv);
                    if (diag_out && row == col)
                        diag_out[(long)zb * diag_bs + (long)row * diag_ld]
                            = (__bf16)acc[mi][ni][r];
                }
            }
        }
    }
}

// ---------------------------------------------------------------------------
// Launch
// ---------------------------------------------------------------------------
extern "C" void kernel_launch(void* const* d_in, const int* in_sizes, int n_in,
                              void* d_out, int out_size, void* d_ws, size_t ws_size,
                              hipStream_t stream)
{
    const int*           code      = (const int*)d_in[0];
    const int*           pos       = (const int*)d_in[1];
    const unsigned char* attn      = (const unsigned char*)d_in[2];
    const float*         emb_table = (const float*)d_in[3];
    const float*         w_lin     = (const float*)d_in[4];
    const float*         b_lin     = (const float*)d_in[5];
    float*               out       = (float*)d_out;

    char* ws = (char*)d_ws;
    const size_t MATB = (size_t)BB * SS * SS * sizeof(__bf16);   // 2 MB
    __bf16* Mbf  = (__bf16*)(ws);
    __bf16* Mtbf = (__bf16*)(ws + MATB);
    __bf16* Pa   = (__bf16*)(ws + 2 * MATB);
    __bf16* Pb   = (__bf16*)(ws + 3 * MATB);
    __bf16* X    = (__bf16*)(ws + 4 * MATB);                     // [2048][800] bf16

    // 1) M = D^-1 A (+ M^T, + pe col 0 = diag(M))
    build_m_kernel<<<ROWS, 256, 0, stream>>>(attn, Mbf, Mtbf, X + DEMB);

    // 2) embeddings + node averaging -> X cols [0,768), zero pad [784,800)
    emb_avg_kernel<<<ROWS, 256, 0, stream>>>(code, pos, attn, emb_table, X);

    // 3) PE chain: P_i = P_{i-1} * M (B operand = pre-transposed Mt),
    //    diag scattered into X cols 768+step
    const long bsq = (long)SS * SS;
    dim3 gA(SS / 128, SS / 128, BB);       // (4, 4, 4)
    const __bf16* Aprev = Mbf;
    __bf16* pbufs[2] = {Pa, Pb};
    for (int step = 1; step <= KPE - 1; ++step) {
        __bf16* Pn = pbufs[(step - 1) & 1];
        wmma_gemm_kernel<<<gA, 256, 0, stream>>>(
            Aprev, (long)SS, bsq, Mtbf, (long)SS, bsq, Pn, (long)SS, bsq,
            SS, SS, SS, (const float*)nullptr,
            X + DEMB + step, (long)KPAD, (long)SS * KPAD);
        Aprev = Pn;
    }

    // 4) logits = X * w_lin + b_lin   (2048 x 800 x 50265, bf16 WMMA, f32 out)
    dim3 gB(ROWS / 128, (VOCAB + 127) / 128, 1);
    wmma_gemm_kernel<<<gB, 256, 0, stream>>>(
        X, (long)KPAD, 0L, w_lin, (long)VOCAB, 0L, out, (long)VOCAB, 0L,
        VOCAB, KPAD, DCAT, b_lin,
        (__bf16*)nullptr, 0L, 0L);
}
